// BaseLSTM_75050258530685
// MI455X (gfx1250) — compile-verified, matrix operand order checked by reading, not run on hardware
//
#include <hip/hip_runtime.h>
#include <stdint.h>

// Problem constants (from reference)
#define T_STEPS 2048
#define BATCH   32
#define DIM     512
#define OUT_DIM 512
#define NBLK_REC 16                     // recurrence blocks; each owns 32 columns

static const size_t GS = (size_t)T_STEPS * BATCH * DIM;   // word / per-gate elements
#define WSZ (DIM * DIM)                 // one weight matrix
#define HB  (BATCH * DIM)               // one h buffer

typedef __attribute__((ext_vector_type(16))) __bf16 v16bf;
typedef __attribute__((ext_vector_type(8)))  float  v8f;

__device__ __forceinline__ uint32_t pack_bf16(float a, float b) {
  uint32_t ua = __float_as_uint(a);
  uint32_t ub = __float_as_uint(b);
  return ((ua + 0x8000u) >> 16) | ((ub + 0x8000u) & 0xFFFF0000u);
}

// Load a 16(M/N) x 32(K) bf16 WMMA fragment from row-major bf16 src (global).
// Per-lane data = two contiguous 16B chunks -> two global_load_b128.
__device__ __forceinline__ v16bf load_frag_bf(const __bf16* __restrict__ src, int ld,
                                              int row0, int k0, int lane) {
  const int r  = lane & 15;
  const int hi = lane >> 4;
  const __bf16* p = src + (size_t)(row0 + r) * ld + k0 + hi * 8;
  union { v16bf v; int4 q[2]; } f;
  f.q[0] = *(const int4*)(p);
  f.q[1] = *(const int4*)(p + 16);
  return f.v;
}

__device__ __forceinline__ v8f wmma_bf16(v16bf a, v16bf b, v8f c) {
  return __builtin_amdgcn_wmma_f32_16x16x32_bf16(false, a, false, b,
                                                 (short)0, c, false, false);
}

__device__ __forceinline__ float sigmoidf(float x) {
  return 1.0f / (1.0f + __expf(-x));
}

// ---------------------------------------------------------------------------
// Kernel 0: fp32 -> bf16 streaming conversion (8 floats -> one uint4 store).
// ---------------------------------------------------------------------------
__global__ void __launch_bounds__(256)
cvt_kernel(const float* __restrict__ src, uint32_t* __restrict__ dst, int n4) {
  const int stride = gridDim.x * blockDim.x;
  for (int j = blockIdx.x * blockDim.x + threadIdx.x; j < n4; j += stride) {
    const float4 a = ((const float4*)src)[2 * j];
    const float4 b = ((const float4*)src)[2 * j + 1];
    uint4 o;
    o.x = pack_bf16(a.x, a.y);
    o.y = pack_bf16(a.z, a.w);
    o.z = pack_bf16(b.x, b.y);
    o.w = pack_bf16(b.z, b.w);
    ((uint4*)dst)[j] = o;
  }
}

// ---------------------------------------------------------------------------
// Kernel 1: hoisted input projections with bias folding (all-bf16 operands).
//   G0 = word@Wi^T + 2*bi ; G1 = word@Wz^T + bz + bi ; G2 = word@Wo^T + bo + bi
// The block's 32-column weight slab (32x512 bf16 = 32KB) is staged in LDS once
// and B fragments come from ds_load_b128; A fragments stream from global.
// Each wave: 32(M) x 32(N) tile -> 4 wmma per K-step (4 global + 4 ds loads).
// Grid: (256 row groups of 8 waves x 32 rows, 16 col groups of 32, 3 gates).
// ---------------------------------------------------------------------------
__global__ void __launch_bounds__(256)
xproj_kernel(const __bf16* __restrict__ wordb,
             const __bf16* __restrict__ Wib,
             const __bf16* __restrict__ Wzb,
             const __bf16* __restrict__ Wob,
             const float* __restrict__ bi, const float* __restrict__ bz,
             const float* __restrict__ bo,
             float* __restrict__ G) {
  __shared__ __align__(16) __bf16 Bs[32 * DIM];   // 32 KB weight slab
  const int lane = threadIdx.x & 31;
  const int wave = threadIdx.x >> 5;
  const int rowTile = (blockIdx.x * 8 + wave) * 32;   // over T*B = 65536 rows
  const int colBase = blockIdx.y * 32;
  const int g = blockIdx.z;

  const __bf16* W = (g == 0) ? Wib : (g == 1) ? Wzb : Wob;
  float* Gg = G + (size_t)g * GS;

  // Cooperative slab copy: rows [colBase, colBase+32) of W are contiguous.
  {
    const uint4* srcq = (const uint4*)(W + (size_t)colBase * DIM);
    uint4* dstq = (uint4*)Bs;
    for (int i = threadIdx.x; i < 32 * DIM / 8; i += 256)   // 2048 x 16B
      dstq[i] = srcq[i];
  }
  __syncthreads();

  const int r  = lane & 15;
  const int hi = lane >> 4;
  const int nl = lane & 15;

  // B fragment from LDS slab (row-major, ld = DIM) -> two ds_load_b128.
  auto lds_frag = [&](int row0, int k0) -> v16bf {
    union { v16bf v; int4 q[2]; } f;
    const int idx = (row0 + r) * DIM + k0 + hi * 8;
    f.q[0] = *(const int4*)&Bs[idx];
    f.q[1] = *(const int4*)&Bs[idx + 16];
    return f.v;
  };

  v8f acc[2][2] = {};
  for (int k0 = 0; k0 < DIM; k0 += 32) {
    const v16bf a0 = load_frag_bf(wordb, DIM, rowTile,      k0, lane);
    const v16bf a1 = load_frag_bf(wordb, DIM, rowTile + 16, k0, lane);
#pragma unroll
    for (int n = 0; n < 2; ++n) {
      const v16bf b = lds_frag(n * 16, k0);
      acc[0][n] = wmma_bf16(a0, b, acc[0][n]);
      acc[1][n] = wmma_bf16(a1, b, acc[1][n]);
    }
  }

#pragma unroll
  for (int n = 0; n < 2; ++n) {
    const int col = colBase + n * 16 + nl;
    float bias;
    if (g == 0)      bias = 2.0f * bi[col];
    else if (g == 1) bias = bz[col] + bi[col];
    else             bias = bo[col] + bi[col];
#pragma unroll
    for (int mt = 0; mt < 2; ++mt)
#pragma unroll
      for (int rr = 0; rr < 8; ++rr) {
        const int row = rowTile + mt * 16 + rr + hi * 8;
        Gg[(size_t)row * DIM + col] = acc[mt][n][rr] + bias;
      }
  }
}

// ---------------------------------------------------------------------------
// Kernel 2: persistent recurrence. 16 blocks x 256 threads (8 waves).
// Block jb owns columns [jb*32, jb*32+32). Wave w owns K slice [w*64, +64)
// and keeps its Wi B-fragments in registers for all 2048 steps. h is kept
// in bf16 (double buffered in global). The step-t gate pre-activations are
// loaded at the TOP of the step so HBM latency overlaps the WMMA stage, and
// step t+1's lines are prefetched into L2. Partials reduced in LDS; gates
// fused; packed bf16 stores; atomic-counter global barrier per step.
// ---------------------------------------------------------------------------
__global__ void __launch_bounds__(256)
recur_kernel(const float* __restrict__ G,
             const __bf16* __restrict__ Wib,
             __bf16* __restrict__ hbuf,    // [2][BATCH][DIM] bf16, pre-zeroed
             int* __restrict__ counter) {  // pre-zeroed
  __shared__ float part[8][32][32];
  const int lane = threadIdx.x & 31;
  const int wave = threadIdx.x >> 5;
  const int colBase = blockIdx.x * 32;
  const int k0w = wave * 64;

  const float* Gi = G;
  const float* Gz = G + GS;
  const float* Go = G + 2 * GS;

  // Resident weight fragments for this block's column chunk / wave's K slice.
  v16bf Bf[2][2];
#pragma unroll
  for (int nt = 0; nt < 2; ++nt)
#pragma unroll
    for (int kt = 0; kt < 2; ++kt)
      Bf[nt][kt] = load_frag_bf(Wib, DIM, colBase + nt * 16, k0w + kt * 32, lane);

  const int hi = lane >> 4;
  const int nl = lane & 15;

  // Fixed reduce assignments: thread handles elements e0 and e0+256 of the
  // 512 (m, even-n) pairs in this block's 32x32 output.
  const int e0 = threadIdx.x;
  const int e1 = threadIdx.x + 256;
  const int m0 = e0 >> 4, n20 = (e0 & 15) * 2;
  const int m1 = e1 >> 4, n21 = (e1 & 15) * 2;

  for (int t = 0; t < T_STEPS; ++t) {
    const __bf16* hr = hbuf + (size_t)(t & 1) * HB;
    __bf16*       hw = hbuf + (size_t)((t + 1) & 1) * HB;

    // Early G loads for this step (overlap HBM latency with matmul below).
    const size_t idx0 = ((size_t)t * BATCH + m0) * DIM + colBase + n20;
    const size_t idx1 = ((size_t)t * BATCH + m1) * DIM + colBase + n21;
    const float2 gi0 = *(const float2*)&Gi[idx0];
    const float2 gz0 = *(const float2*)&Gz[idx0];
    const float2 go0 = *(const float2*)&Go[idx0];
    const float2 gi1 = *(const float2*)&Gi[idx1];
    const float2 gz1 = *(const float2*)&Gz[idx1];
    const float2 go1 = *(const float2*)&Go[idx1];
    // Pull step t+1's gate lines toward L2.
    if (t + 1 < T_STEPS) {
      __builtin_prefetch(&Gi[idx0 + HB], 0, 1);
      __builtin_prefetch(&Gz[idx0 + HB], 0, 1);
      __builtin_prefetch(&Go[idx0 + HB], 0, 1);
      __builtin_prefetch(&Gi[idx1 + HB], 0, 1);
      __builtin_prefetch(&Gz[idx1 + HB], 0, 1);
      __builtin_prefetch(&Go[idx1 + HB], 0, 1);
    }

    v8f acc[2][2] = {};
#pragma unroll
    for (int kt = 0; kt < 2; ++kt) {
#pragma unroll
      for (int mt = 0; mt < 2; ++mt) {
        const v16bf a = load_frag_bf(hr, DIM, mt * 16, k0w + kt * 32, lane);
#pragma unroll
        for (int nt = 0; nt < 2; ++nt)
          acc[mt][nt] = wmma_bf16(a, Bf[nt][kt], acc[mt][nt]);
      }
    }

#pragma unroll
    for (int mt = 0; mt < 2; ++mt)
#pragma unroll
      for (int nt = 0; nt < 2; ++nt)
#pragma unroll
        for (int rr = 0; rr < 8; ++rr)
          part[wave][mt * 16 + rr + hi * 8][nt * 16 + nl] = acc[mt][nt][rr];

    __syncthreads();

    // Cross-wave K reduction + fused gates; write bf16 pairs (packed dwords).
    uint32_t* hwd = (uint32_t*)hw;
    {
      float s00 = 0.0f, s01 = 0.0f, s10 = 0.0f, s11 = 0.0f;
#pragma unroll
      for (int w = 0; w < 8; ++w) {
        s00 += part[w][m0][n20];
        s01 += part[w][m0][n20 + 1];
        s10 += part[w][m1][n21];
        s11 += part[w][m1][n21 + 1];
      }
      const float h00 = sigmoidf(go0.x + s00) *
                        tanhf(sigmoidf(gi0.x + s00) * sigmoidf(gz0.x + s00));
      const float h01 = sigmoidf(go0.y + s01) *
                        tanhf(sigmoidf(gi0.y + s01) * sigmoidf(gz0.y + s01));
      const float h10 = sigmoidf(go1.x + s10) *
                        tanhf(sigmoidf(gi1.x + s10) * sigmoidf(gz1.x + s10));
      const float h11 = sigmoidf(go1.y + s11) *
                        tanhf(sigmoidf(gi1.y + s11) * sigmoidf(gz1.y + s11));
      hwd[(m0 * DIM + colBase + n20) >> 1] = pack_bf16(h00, h01);
      hwd[(m1 * DIM + colBase + n21) >> 1] = pack_bf16(h10, h11);
    }

    __syncthreads();
    __threadfence();
    if (threadIdx.x == 0) {
      __hip_atomic_fetch_add(counter, 1, __ATOMIC_RELEASE,
                             __HIP_MEMORY_SCOPE_AGENT);
      const int target = NBLK_REC * (t + 1);
      while (__hip_atomic_load(counter, __ATOMIC_ACQUIRE,
                               __HIP_MEMORY_SCOPE_AGENT) < target)
        __builtin_amdgcn_s_sleep(1);
    }
    __syncthreads();
  }
}

// ---------------------------------------------------------------------------
// Kernel 3: output = h_final @ Wy^T + by   (32 x 512, K = 512, bf16 operands)
// ---------------------------------------------------------------------------
__global__ void __launch_bounds__(32)
out_kernel(const __bf16* __restrict__ hfin,
           const __bf16* __restrict__ Wyb,
           const float* __restrict__ by,
           float* __restrict__ out) {
  const int lane = threadIdx.x & 31;
  const int rowTile = blockIdx.x * 16;
  const int colTile = blockIdx.y * 16;
  v8f acc = {};
  for (int k0 = 0; k0 < DIM; k0 += 32) {
    const v16bf a = load_frag_bf(hfin, DIM, rowTile, k0, lane);
    const v16bf b = load_frag_bf(Wyb,  DIM, colTile, k0, lane);
    acc = wmma_bf16(a, b, acc);
  }
  const int hi = lane >> 4, nl = lane & 15;
  const int col = colTile + nl;
  const float bias = by[col];
#pragma unroll
  for (int rr = 0; rr < 8; ++rr)
    out[(size_t)(rowTile + rr + hi * 8) * OUT_DIM + col] = acc[rr] + bias;
}

extern "C" void kernel_launch(void* const* d_in, const int* in_sizes, int n_in,
                              void* d_out, int out_size, void* d_ws, size_t ws_size,
                              hipStream_t stream) {
  (void)in_sizes; (void)n_in; (void)out_size; (void)ws_size;
  // setup_inputs order: word, Wf, bf, Wi, bi, Wz, bz, Wo, bo, Wy, by
  const float* word = (const float*)d_in[0];
  const float* Wi   = (const float*)d_in[3];
  const float* bi   = (const float*)d_in[4];
  const float* Wz   = (const float*)d_in[5];
  const float* bz   = (const float*)d_in[6];
  const float* Wo   = (const float*)d_in[7];
  const float* bo   = (const float*)d_in[8];
  const float* Wy   = (const float*)d_in[9];
  const float* by   = (const float*)d_in[10];

  // Workspace layout:
  //   bf16 wordb[GS] | bf16 Wib/Wzb/Wob/Wyb[4*WSZ] | f32 G[3*GS]
  //   | bf16 hbuf[2*HB] | int counter
  __bf16* wordb = (__bf16*)d_ws;
  __bf16* Wib   = wordb + GS;
  __bf16* Wzb   = Wib + WSZ;
  __bf16* Wob   = Wzb + WSZ;
  __bf16* Wyb   = Wob + WSZ;
  float*  G     = (float*)(Wyb + WSZ);
  __bf16* hbuf  = (__bf16*)(G + 3 * GS);
  int* counter  = (int*)(hbuf + 2 * HB);

  // h0 = 0 and barrier counter = 0 (graph-capturable async memset).
  hipMemsetAsync(hbuf, 0, (size_t)2 * HB * sizeof(__bf16) + 256, stream);

  // Phase 0: one-time fp32 -> bf16 conversion (streaming, HBM-bound).
  cvt_kernel<<<2048, 256, 0, stream>>>(word, (uint32_t*)wordb, (int)(GS / 8));
  cvt_kernel<<<128, 256, 0, stream>>>(Wi, (uint32_t*)Wib, WSZ / 8);
  cvt_kernel<<<128, 256, 0, stream>>>(Wz, (uint32_t*)Wzb, WSZ / 8);
  cvt_kernel<<<128, 256, 0, stream>>>(Wo, (uint32_t*)Wob, WSZ / 8);
  cvt_kernel<<<128, 256, 0, stream>>>(Wy, (uint32_t*)Wyb, WSZ / 8);

  // Phase 1: all input projections as one parallel WMMA GEMM pass.
  xproj_kernel<<<dim3(256, 16, 3), 256, 0, stream>>>(wordb, Wib, Wzb, Wob,
                                                     bi, bz, bo, G);

  // Phase 2: sequential recurrence, 16 persistent blocks.
  recur_kernel<<<dim3(NBLK_REC), 256, 0, stream>>>(G, Wib, hbuf, counter);

  // Phase 3: final projection. Final h lives in hbuf[T_STEPS & 1].
  const __bf16* hfin = hbuf + (size_t)(T_STEPS & 1) * HB;
  out_kernel<<<dim3(2, 32), 32, 0, stream>>>(hfin, Wyb, by, (float*)d_out);
}
